// RecurrentEmbeddingModel_55465207660676
// MI455X (gfx1250) — matrix operand order repaired and checked by reading, not compiled, not run on hardware
//
#include <hip/hip_runtime.h>

typedef __attribute__((ext_vector_type(16))) __bf16 v16bf;
typedef __attribute__((ext_vector_type(8)))  float  v8f;
typedef __attribute__((address_space(3)))    float  ldsf;

#define HDIM 32
#define TMAX 256
#define SEQ_PER_WAVE 16
#define WAVES_PB 4          // 128 threads/block, 64 KB dynamic LDS
#define NSEQ (32768 * 4)

__device__ __forceinline__ float fast_tanh(float y) {
#if __has_builtin(__builtin_amdgcn_tanhf)
    return __builtin_amdgcn_tanhf(y);            // v_tanh_f32
#elif __has_builtin(__builtin_amdgcn_rcpf)
    float e = __expf(2.f * y);                   // v_exp_f32
    return (e - 1.f) * __builtin_amdgcn_rcpf(e + 1.f);
#else
    float e = __expf(2.f * y);
    return (e - 1.f) / (e + 1.f);
#endif
}

// ---------- length-bucketing: counting sort by length (257 bins) ----------
__global__ __launch_bounds__(256)
void hist_kernel(const int* __restrict__ l, int* __restrict__ hist)
{
    int i = blockIdx.x * blockDim.x + threadIdx.x;
    if (i < NSEQ) {
        int len = min(max(l[i], 0), TMAX);
        atomicAdd(&hist[len], 1);
    }
}

__global__ __launch_bounds__(512)
void scan_kernel(const int* __restrict__ hist, int* __restrict__ offs)
{
    __shared__ int s[TMAX + 1];
    if (threadIdx.x <= TMAX) s[threadIdx.x] = hist[threadIdx.x];
    __syncthreads();
    if (threadIdx.x == 0) {
        int acc = 0;
        for (int b = 0; b <= TMAX; ++b) { int c = s[b]; s[b] = acc; acc += c; }
    }
    __syncthreads();
    if (threadIdx.x <= TMAX) offs[threadIdx.x] = s[threadIdx.x];
}

__global__ __launch_bounds__(256)
void scatter_kernel(const int* __restrict__ l, int* __restrict__ offs,
                    int* __restrict__ perm)
{
    int i = blockIdx.x * blockDim.x + threadIdx.x;
    if (i < NSEQ) {
        int len = min(max(l[i], 0), TMAX);
        int pos = atomicAdd(&offs[len], 1);
        perm[pos] = i;      // bin-internal order arbitrary; output invariant
    }
}

// ---------------- Kernel 1: batched tanh-RNN via WMMA ----------------
// One wave handles 16 length-bucketed sequences. Hidden state (32 x 16)
// lives in two f32 16x16 C/D-layout tiles. Per step:
//   B(bf16 32x16) <- repack(hidden)  [bf16 packs of h0|h1, no shuffles]
//   acc = v_wmma_f32_16x16x32_bf16(A=W_hh tile, B, C=bias)   x2 tiles
//   hid = tanh(acc + x*w_ih)   (masked variant only for mixed-length waves)
__global__ __launch_bounds__(WAVES_PB * 32)
void rnn_wmma_kernel(const float* __restrict__ hseq,
                     const int*   __restrict__ lens,
                     const int*   __restrict__ perm,
                     const float* __restrict__ W_ih,
                     const float* __restrict__ W_hh,
                     const float* __restrict__ b_ih,
                     const float* __restrict__ b_hh,
                     float*       __restrict__ feat)   // (N, R*H) = (N,128)
{
    extern __shared__ float smem[];                     // WAVES_PB*16*256 floats
    const int lane = threadIdx.x & 31;
    const int wave = threadIdx.x >> 5;
    const int seqbase = (blockIdx.x * WAVES_PB + wave) * SEQ_PER_WAVE;

    unsigned lds_base = (unsigned)(size_t)smem
                      + (unsigned)(wave * SEQ_PER_WAVE * TMAX) * 4u;

    // ---- stage 16 (scattered) 1KB sequences via async global->LDS ----
    // Each instruction moves 32B from each of the 16 rows (2 lanes/row).
    {
        int      cs    = lane >> 1;                       // column 0..15
        int      ps    = perm[seqbase + cs];              // source sequence
        unsigned sub   = (unsigned)(lane & 1) * 16u;
        unsigned goff0 = (unsigned)ps * (TMAX * 4u) + sub;
        unsigned loff0 = lds_base + (unsigned)cs * (TMAX * 4u) + sub;
        unsigned long long gbase = (unsigned long long)(size_t)hseq;
        #pragma unroll
        for (int i = 0; i < 32; ++i) {
            unsigned go = goff0 + (unsigned)i * 32u;
            unsigned lo = loff0 + (unsigned)i * 32u;
            asm volatile("global_load_async_to_lds_b128 %0, %1, %2"
                         :: "v"(lo), "v"(go), "s"(gbase) : "memory");
        }
    }

    // ---- per-lane constants while the async copy is in flight ----
    const int m       = lane & 15;       // A-layout row / D-layout column
    const int halfsel = lane >> 4;       // lane half

    const int pseq = perm[seqbase + m];  // this lane's sequence (both halves)
    int len = lens[pseq];
    len = min(max(len, 0), TMAX);
    int maxlen = len, minlen = len;
    #pragma unroll
    for (int o = 8; o >= 1; o >>= 1) {
        maxlen = max(maxlen, __shfl_xor(maxlen, o, 32));
        minlen = min(minlen, __shfl_xor(minlen, o, 32));
    }
    maxlen = __builtin_amdgcn_readfirstlane(maxlen);    // scalar loop bounds
    minlen = __builtin_amdgcn_readfirstlane(minlen);

    // A tiles: A[m,k] = W_hh[m(+16), k]; 16-bit A layout:
    // lanes 0-15 hold K {0..7,16..23}, lanes 16-31 hold K {8..15,24..31}
    v16bf A0, A1;
    #pragma unroll
    for (int v = 0; v < 8; ++v) {
        int kb = ((v < 4) ? 2 * v : 16 + 2 * (v - 4)) + 8 * halfsel;
        A0[2*v]   = (__bf16)W_hh[m * HDIM + kb];
        A0[2*v+1] = (__bf16)W_hh[m * HDIM + kb + 1];
        A1[2*v]   = (__bf16)W_hh[(m + 16) * HDIM + kb];
        A1[2*v+1] = (__bf16)W_hh[(m + 16) * HDIM + kb + 1];
    }
    // D-layout row constants: reg r -> row r + 8*halfsel (+16 for tile1)
    v8f   cbias0, cbias1;
    float wih0[8], wih1[8];
    #pragma unroll
    for (int r = 0; r < 8; ++r) {
        int r0 = r + 8 * halfsel, r1 = r0 + 16;
        cbias0[r] = b_ih[r0] + b_hh[r0];
        cbias1[r] = b_ih[r1] + b_hh[r1];
        wih0[r] = W_ih[r0];
        wih1[r] = W_ih[r1];
    }

    float h0[8], h1[8];
    #pragma unroll
    for (int r = 0; r < 8; ++r) { h0[r] = 0.f; h1[r] = 0.f; }

    // Explicit LDS (AS3) pointer -> ds_load_b32 (DScnt only), never FLAT.
    const ldsf* xrow =
        (const ldsf*)(unsigned)(lds_base + (unsigned)(m * TMAX) * 4u);

    asm volatile("s_wait_asynccnt 0" ::: "memory");

    if (minlen == maxlen) {
        // --------- fast path: uniform lengths (most waves after sorting) ---------
        #pragma unroll 1
        for (int t = 0; t < maxlen; ++t) {
            float x = xrow[len - 1 - t];               // always valid: t < len
            v16bf B;
            #pragma unroll
            for (int i = 0; i < 8; ++i) {
                B[i]     = (__bf16)h0[i];
                B[8 + i] = (__bf16)h1[i];
            }
            v8f acc0 = __builtin_amdgcn_wmma_f32_16x16x32_bf16(
                false, A0, false, B, (short)0, cbias0, false, false);
            v8f acc1 = __builtin_amdgcn_wmma_f32_16x16x32_bf16(
                false, A1, false, B, (short)0, cbias1, false, false);
            #pragma unroll
            for (int r = 0; r < 8; ++r) {
                h0[r] = fast_tanh(fmaf(x, wih0[r], acc0[r]));
                h1[r] = fast_tanh(fmaf(x, wih1[r], acc1[r]));
            }
        }
    } else {
        // --------- masked path: mixed-length (bin-boundary) waves ---------
        #pragma unroll 1
        for (int t = 0; t < maxlen; ++t) {
            int   idx = min(max(len - 1 - t, 0), TMAX - 1);
            float x   = xrow[idx];
            bool  act = t < len;
            v16bf B;
            #pragma unroll
            for (int i = 0; i < 8; ++i) {
                B[i]     = (__bf16)h0[i];
                B[8 + i] = (__bf16)h1[i];
            }
            v8f acc0 = __builtin_amdgcn_wmma_f32_16x16x32_bf16(
                false, A0, false, B, (short)0, cbias0, false, false);
            v8f acc1 = __builtin_amdgcn_wmma_f32_16x16x32_bf16(
                false, A1, false, B, (short)0, cbias1, false, false);
            #pragma unroll
            for (int r = 0; r < 8; ++r) {
                float t0 = fast_tanh(fmaf(x, wih0[r], acc0[r]));
                float t1 = fast_tanh(fmaf(x, wih1[r], acc1[r]));
                h0[r] = act ? t0 : h0[r];
                h1[r] = act ? t1 : h1[r];
            }
        }
    }

    // Scatter final hidden into feat (N, R*32): row = n, col = route*32 + j
    size_t fb = (size_t)(pseq >> 2) * 128 + (size_t)(pseq & 3) * 32;
    #pragma unroll
    for (int r = 0; r < 8; ++r) {
        feat[fb + r + 8 * halfsel]      = h0[r];
        feat[fb + r + 8 * halfsel + 16] = h1[r];
    }
}

// ---------------- Kernel 2: tiny MLP, one wave per batch row ----------------
__global__ __launch_bounds__(256)
void mlp_kernel(const float* __restrict__ feat,
                const float* __restrict__ W1, const float* __restrict__ b1,
                const float* __restrict__ W2, const float* __restrict__ b2,
                const float* __restrict__ W3, const float* __restrict__ b3,
                float* __restrict__ out)
{
    __shared__ float sW1[128 * 32];
    __shared__ float sW2[32 * 32];
    __shared__ float sW3[32 * 4];
    __shared__ float sb1[32], sb2[32], sb3[4];
    for (int i = threadIdx.x; i < 128 * 32; i += blockDim.x) sW1[i] = W1[i];
    for (int i = threadIdx.x; i < 32 * 32;  i += blockDim.x) sW2[i] = W2[i];
    for (int i = threadIdx.x; i < 128;      i += blockDim.x) sW3[i] = W3[i];
    if (threadIdx.x < 32) { sb1[threadIdx.x] = b1[threadIdx.x];
                            sb2[threadIdx.x] = b2[threadIdx.x]; }
    if (threadIdx.x < 4)    sb3[threadIdx.x] = b3[threadIdx.x];
    __syncthreads();

    const int lane = threadIdx.x & 31;
    const int wave = threadIdx.x >> 5;
    const int row  = blockIdx.x * 8 + wave;

    const float* f = feat + (size_t)row * 128;
    float fl[4];
    #pragma unroll
    for (int q = 0; q < 4; ++q) fl[q] = f[lane + 32 * q];

    float z1 = sb1[lane];
    #pragma unroll 8
    for (int k = 0; k < 128; ++k) {
        float xk = __shfl(fl[k >> 5], k & 31, 32);
        z1 = fmaf(xk, sW1[k * 32 + lane], z1);
    }
    z1 = fmaxf(z1, 0.f);

    float z2 = sb2[lane];
    #pragma unroll
    for (int k = 0; k < 32; ++k) {
        float xk = __shfl(z1, k, 32);
        z2 = fmaf(xk, sW2[k * 32 + lane], z2);
    }
    z2 = fmaxf(z2, 0.f);

    float z3 = (lane < 4) ? sb3[lane] : 0.f;
    #pragma unroll
    for (int k = 0; k < 32; ++k) {
        float xk = __shfl(z2, k, 32);
        z3 = fmaf(xk, sW3[k * 4 + (lane & 3)], z3);
    }
    if (lane < 4) out[(size_t)row * 4 + lane] = z3;
}

extern "C" void kernel_launch(void* const* d_in, const int* in_sizes, int n_in,
                              void* d_out, int out_size, void* d_ws, size_t ws_size,
                              hipStream_t stream)
{
    const float* h    = (const float*)d_in[0];
    const int*   l    = (const int*)  d_in[1];
    const float* W_ih = (const float*)d_in[2];
    const float* W_hh = (const float*)d_in[3];
    const float* b_ih = (const float*)d_in[4];
    const float* b_hh = (const float*)d_in[5];
    const float* W1   = (const float*)d_in[6];
    const float* b1   = (const float*)d_in[7];
    const float* W2   = (const float*)d_in[8];
    const float* b2   = (const float*)d_in[9];
    const float* W3   = (const float*)d_in[10];
    const float* b3   = (const float*)d_in[11];

    const int N = 32768;

    // Workspace layout: feat (16 MB) | perm (512 KB) | hist (257) | offs (257)
    char*  ws   = (char*)d_ws;
    float* feat = (float*)ws;
    int*   perm = (int*)(ws + (size_t)N * 128 * sizeof(float));
    int*   hist = (int*)((char*)perm + (size_t)NSEQ * sizeof(int));
    int*   offs = hist + 512;

    hipMemsetAsync(hist, 0, (TMAX + 1) * sizeof(int), stream);
    hist_kernel<<<(NSEQ + 255) / 256, 256, 0, stream>>>(l, hist);
    scan_kernel<<<1, 512, 0, stream>>>(hist, offs);
    scatter_kernel<<<(NSEQ + 255) / 256, 256, 0, stream>>>(l, offs, perm);

    dim3 grid1(NSEQ / (WAVES_PB * SEQ_PER_WAVE));
    size_t shmem = (size_t)WAVES_PB * SEQ_PER_WAVE * TMAX * sizeof(float); // 64 KB
    rnn_wmma_kernel<<<grid1, WAVES_PB * 32, shmem, stream>>>(
        h, l, perm, W_ih, W_hh, b_ih, b_hh, feat);

    mlp_kernel<<<N / 8, 256, 0, stream>>>(feat, W1, b1, W2, b2, W3, b3,
                                          (float*)d_out);
}
